// DuelingDQN_52776558133770
// MI455X (gfx1250) — matrix operand order repaired and checked by reading, not compiled, not run on hardware
//
#include <hip/hip_runtime.h>
#include <math.h>

// ---------------- problem constants ----------------
#define EPSV 1e-5f
#define BB 32
#define SS 512
#define FF 100
#define HH 512

typedef __attribute__((ext_vector_type(16))) __bf16 v16bf;
typedef __attribute__((ext_vector_type(8)))  __bf16 v8bf;
typedef __attribute__((ext_vector_type(8)))  float  v8f;
typedef __attribute__((ext_vector_type(4)))  float  v4f;

__device__ __forceinline__ __bf16 f2bf(float f) { return (__bf16)f; }
__device__ __forceinline__ float  bf2f(__bf16 b) { return (float)b; }
__device__ __forceinline__ float  sigm(float x) { return 1.0f / (1.0f + __expf(-x)); }

__device__ __forceinline__ v16bf cat16(v8bf lo, v8bf hi) {
  return __builtin_shufflevector(lo, hi, 0, 1, 2, 3, 4, 5, 6, 7, 8, 9, 10, 11, 12, 13, 14, 15);
}

// ---------------- weight pack: B (KxN f32, row-major) -> Bp (Np x Kp bf16, zero-padded) ----
__global__ void pack_bt(const float* __restrict__ B, int K, int N,
                        __bf16* __restrict__ Bp, int Kp, int Np) {
  size_t total = (size_t)Np * Kp;
  for (size_t i = (size_t)blockIdx.x * blockDim.x + threadIdx.x; i < total;
       i += (size_t)gridDim.x * blockDim.x) {
    int n = (int)(i / Kp), k = (int)(i % Kp);
    float v = (n < N && k < K) ? B[(size_t)k * N + n] : 0.0f;
    Bp[i] = f2bf(v);
  }
}

// ---------------- GEMM: C(MxN) = A(MxKp f32) @ Bp (+bias)(+relu)(+acc) ----------------
// M multiple of 16 at all call sites. Wave owns a 16x64 tile (4 WMMAs share one A frag).
// k-loop is software-pipelined: next fragments are loaded before current WMMAs issue.
#define GF_BIAS 1
#define GF_RELU 2
#define GF_ACC  4

struct AFrag { v4f a0, a1, a2, a3; };

__global__ void __launch_bounds__(256) gemm_bf16_wmma(
    const float* __restrict__ A, int lda,
    const __bf16* __restrict__ Bp, int Kp,
    const float* __restrict__ bias,
    float* __restrict__ C, int ldc,
    int M, int N, int flags) {
  const int lane = threadIdx.x & 31;
  const int g    = lane >> 4;
  const int mn   = lane & 15;
  const int wave = threadIdx.x >> 5;
  const int wpb  = blockDim.x >> 5;
  const int tm   = M >> 4;
  const int tn4  = (N + 63) >> 6;
  const int ntiles = tm * tn4;
  for (int tile = blockIdx.x * wpb + wave; tile < ntiles; tile += gridDim.x * wpb) {
    const int m0 = (tile / tn4) << 4;
    const int n0 = (tile % tn4) << 6;
    const float*  Arow = A + (size_t)(m0 + mn) * lda + (g << 3);
    const __bf16* Bcol = Bp + (size_t)(n0 + mn) * Kp + (g << 3);

    auto loadA = [&](int k0) -> AFrag {
      const v4f* ap = (const v4f*)(Arow + k0);
      AFrag f; f.a0 = ap[0]; f.a1 = ap[1]; f.a2 = ap[4]; f.a3 = ap[5];
      return f;
    };
    auto loadB = [&](int nt, int k0) -> v16bf {
      const __bf16* bp = Bcol + (size_t)(nt << 4) * Kp + k0;
      return cat16(*(const v8bf*)bp, *(const v8bf*)(bp + 16));
    };
    auto cvtA = [&](AFrag f) -> v16bf {
      v16bf af;
#pragma unroll
      for (int e = 0; e < 4; ++e) {
        af[e]      = f2bf(f.a0[e]);
        af[4 + e]  = f2bf(f.a1[e]);
        af[8 + e]  = f2bf(f.a2[e]);
        af[12 + e] = f2bf(f.a3[e]);
      }
      return af;
    };

    v8f acc0 = {}, acc1 = {}, acc2 = {}, acc3 = {};
    AFrag fa  = loadA(0);
    v16bf fb0 = loadB(0, 0), fb1 = loadB(1, 0), fb2 = loadB(2, 0), fb3 = loadB(3, 0);
    for (int k0 = 0; k0 < Kp; k0 += 32) {
      int kn = k0 + 32;
      if (kn >= Kp) kn = 0;  // branchless wrap: one redundant load on the last iteration
      AFrag na  = loadA(kn);
      v16bf nb0 = loadB(0, kn), nb1 = loadB(1, kn), nb2 = loadB(2, kn), nb3 = loadB(3, kn);
      v16bf af = cvtA(fa);
      acc0 = __builtin_amdgcn_wmma_f32_16x16x32_bf16(false, af, false, fb0, (short)0, acc0, false, false);
      acc1 = __builtin_amdgcn_wmma_f32_16x16x32_bf16(false, af, false, fb1, (short)0, acc1, false, false);
      acc2 = __builtin_amdgcn_wmma_f32_16x16x32_bf16(false, af, false, fb2, (short)0, acc2, false, false);
      acc3 = __builtin_amdgcn_wmma_f32_16x16x32_bf16(false, af, false, fb3, (short)0, acc3, false, false);
      fa = na; fb0 = nb0; fb1 = nb1; fb2 = nb2; fb3 = nb3;
    }
#pragma unroll
    for (int nt = 0; nt < 4; ++nt) {
      const int col = n0 + (nt << 4) + mn;
      if (col < N) {
        v8f acc = (nt == 0) ? acc0 : (nt == 1) ? acc1 : (nt == 2) ? acc2 : acc3;
#pragma unroll
        for (int v = 0; v < 8; ++v) {
          const int row = m0 + v + (g << 3);
          size_t idx = (size_t)row * ldc + col;
          float f = acc[v];
          if (flags & GF_ACC)  f += C[idx];
          if (flags & GF_BIAS) f += bias[col];
          if (flags & GF_RELU) f = fmaxf(f, 0.0f);
          C[idx] = f;
        }
      }
    }
  }
}

// ---------------- bidirectional persistent GRU scan ----------------
// grid=2: block 0 = forward, block 1 = backward (independent, concurrent).
// h (32x512 bf16) in LDS; wave w owns h-columns [16w,16w+16) and keeps the r/z/n
// pre-activations for those columns in 6 WMMA accumulators. B fragments come from the
// packed bf16 Whh (L2-resident); the k-loop is software-pipelined, and next step's xp
// gate inputs are prefetched (global_prefetch) to hide HBM latency on the scan path.
__global__ void __launch_bounds__(1024) gru_scan_bi(
    const float* __restrict__ xpf, const float* __restrict__ xpb,
    const __bf16* __restrict__ Whpf, const __bf16* __restrict__ Whpb,
    const float* __restrict__ bhhf, const float* __restrict__ bhhb,
    float* __restrict__ hs, int S_len) {
  const int reverse  = blockIdx.x;
  const float*  xp   = reverse ? xpb : xpf;
  const __bf16* Whp  = reverse ? Whpb : Whpf;
  const float*  bhh  = reverse ? bhhb : bhhf;
  const int     coff = reverse ? HH : 0;

  __shared__ __bf16 hsh[BB][HH];
  const int lane = threadIdx.x & 31;
  const int g    = lane >> 4;
  const int mn   = lane & 15;
  const int wave = threadIdx.x >> 5;
  const int nc0  = wave << 4;
  for (int i = threadIdx.x; i < BB * HH; i += blockDim.x)
    (&hsh[0][0])[i] = f2bf(0.0f);
  __syncthreads();

  const int colL = nc0 + mn;
  const float br  = bhh[colL];
  const float bz  = bhh[HH + colL];
  const float bnn = bhh[2 * HH + colL];

  auto loadAh = [&](int mt, int k0) -> v16bf {
    const __bf16* aptr = &hsh[(mt << 4) + mn][k0 + (g << 3)];
    return cat16(*(const v8bf*)aptr, *(const v8bf*)(aptr + 16));
  };
  auto loadBh = [&](int third, int k0) -> v16bf {
    const __bf16* bp = Whp + (size_t)(third * HH + colL) * HH + k0 + (g << 3);
    return cat16(*(const v8bf*)bp, *(const v8bf*)(bp + 16));
  };

  for (int step = 0; step < S_len; ++step) {
    const int t = reverse ? (S_len - 1 - step) : step;

    // prefetch next step's gate inputs (register-free latency hiding)
    const int tnx = reverse ? (t ? t - 1 : 0) : (t + 1 < S_len ? t + 1 : t);
#pragma unroll
    for (int mt = 0; mt < 2; ++mt)
#pragma unroll
      for (int v = 0; v < 8; ++v) {
        const int row = (mt << 4) + v + (g << 3);
        const size_t xb = ((size_t)(row * S_len + tnx)) * (3 * HH);
        __builtin_prefetch(&xp[xb + colL], 0, 1);
        __builtin_prefetch(&xp[xb + HH + colL], 0, 1);
        __builtin_prefetch(&xp[xb + 2 * HH + colL], 0, 1);
      }

    v8f zero = {};
    v8f acc[3][2];
#pragma unroll
    for (int i = 0; i < 3; ++i) { acc[i][0] = zero; acc[i][1] = zero; }

    v16bf a0 = loadAh(0, 0), a1 = loadAh(1, 0);
    v16bf b0 = loadBh(0, 0), b1 = loadBh(1, 0), b2 = loadBh(2, 0);
    for (int k0 = 0; k0 < HH; k0 += 32) {
      const int kn = (k0 + 32) & (HH - 1);  // branchless wrap
      v16bf na0 = loadAh(0, kn), na1 = loadAh(1, kn);
      v16bf nb0 = loadBh(0, kn), nb1 = loadBh(1, kn), nb2 = loadBh(2, kn);
      acc[0][0] = __builtin_amdgcn_wmma_f32_16x16x32_bf16(false, a0, false, b0, (short)0, acc[0][0], false, false);
      acc[0][1] = __builtin_amdgcn_wmma_f32_16x16x32_bf16(false, a1, false, b0, (short)0, acc[0][1], false, false);
      acc[1][0] = __builtin_amdgcn_wmma_f32_16x16x32_bf16(false, a0, false, b1, (short)0, acc[1][0], false, false);
      acc[1][1] = __builtin_amdgcn_wmma_f32_16x16x32_bf16(false, a1, false, b1, (short)0, acc[1][1], false, false);
      acc[2][0] = __builtin_amdgcn_wmma_f32_16x16x32_bf16(false, a0, false, b2, (short)0, acc[2][0], false, false);
      acc[2][1] = __builtin_amdgcn_wmma_f32_16x16x32_bf16(false, a1, false, b2, (short)0, acc[2][1], false, false);
      a0 = na0; a1 = na1; b0 = nb0; b1 = nb1; b2 = nb2;
    }

    float newv[16];
#pragma unroll
    for (int mt = 0; mt < 2; ++mt)
#pragma unroll
      for (int v = 0; v < 8; ++v) {
        const int row = (mt << 4) + v + (g << 3);
        const size_t xb = ((size_t)(row * S_len + t)) * (3 * HH);
        float xr = xp[xb + colL];
        float xz = xp[xb + HH + colL];
        float xn = xp[xb + 2 * HH + colL];
        float r = sigm(xr + acc[0][mt][v] + br);
        float z = sigm(xz + acc[1][mt][v] + bz);
        float n = tanhf(xn + r * (acc[2][mt][v] + bnn));
        float hold = bf2f(hsh[row][colL]);
        float hnew = (1.0f - z) * n + z * hold;
        newv[(mt << 3) + v] = hnew;
        hs[((size_t)(row * S_len + t)) * (2 * HH) + coff + colL] = hnew;
      }
    __syncthreads();
#pragma unroll
    for (int mt = 0; mt < 2; ++mt)
#pragma unroll
      for (int v = 0; v < 8; ++v) {
        const int row = (mt << 4) + v + (g << 3);
        hsh[row][colL] = f2bf(newv[(mt << 3) + v]);
      }
    __syncthreads();
  }
}

// ---------------- BN over (B, C, S), axes (0,2) ----------------
__global__ void bn_stats(const float* __restrict__ X, int C,
                         float* __restrict__ mean, float* __restrict__ rstd) {
  __shared__ float s1[256], s2[256];
  const int c = blockIdx.x;
  const int n = BB * SS;
  float a = 0.0f, b = 0.0f;
  for (int i = threadIdx.x; i < n; i += 256) {
    int bi = i / SS, si = i - bi * SS;
    float v = X[(size_t)bi * C * SS + (size_t)c * SS + si];
    a += v; b += v * v;
  }
  s1[threadIdx.x] = a; s2[threadIdx.x] = b; __syncthreads();
  for (int o = 128; o > 0; o >>= 1) {
    if (threadIdx.x < o) { s1[threadIdx.x] += s1[threadIdx.x + o]; s2[threadIdx.x] += s2[threadIdx.x + o]; }
    __syncthreads();
  }
  if (threadIdx.x == 0) {
    float m = s1[0] / n;
    float var = s2[0] / n - m * m;
    mean[c] = m;
    rstd[c] = rsqrtf(var + EPSV);
  }
}

__global__ void bn_apply(const float* __restrict__ X, const float* __restrict__ mean,
                         const float* __restrict__ rstd, const float* __restrict__ gamma,
                         const float* __restrict__ beta, float* __restrict__ Y, int C) {
  size_t total = (size_t)BB * C * SS;
  for (size_t i = (size_t)blockIdx.x * blockDim.x + threadIdx.x; i < total;
       i += (size_t)gridDim.x * blockDim.x) {
    int c = (int)((i / SS) % C);
    Y[i] = (X[i] - mean[c]) * rstd[c] * gamma[c] + beta[c];
  }
}

// ---------------- attention softmax-pool over sequence ----------------
__global__ void attn_pool(const float* __restrict__ hsbuf, const float* __restrict__ Watt,
                          float* __restrict__ pooled) {
  __shared__ float logit[SS];
  __shared__ float red[256];
  const int b = blockIdx.x;
  for (int s = threadIdx.x; s < SS; s += blockDim.x) {
    const float* row = hsbuf + ((size_t)(b * SS + s)) * (2 * HH);
    float acc = 0.0f;
    for (int d = 0; d < 2 * HH; ++d) acc += row[d] * Watt[d];
    logit[s] = acc;
  }
  __syncthreads();
  float m = -1e30f;
  for (int s = threadIdx.x; s < SS; s += blockDim.x) m = fmaxf(m, logit[s]);
  red[threadIdx.x] = m; __syncthreads();
  for (int o = 128; o > 0; o >>= 1) {
    if (threadIdx.x < o) red[threadIdx.x] = fmaxf(red[threadIdx.x], red[threadIdx.x + o]);
    __syncthreads();
  }
  m = red[0]; __syncthreads();
  float sum = 0.0f;
  for (int s = threadIdx.x; s < SS; s += blockDim.x) {
    float e = __expf(logit[s] - m);
    logit[s] = e; sum += e;
  }
  red[threadIdx.x] = sum; __syncthreads();
  for (int o = 128; o > 0; o >>= 1) {
    if (threadIdx.x < o) red[threadIdx.x] += red[threadIdx.x + o];
    __syncthreads();
  }
  const float inv = 1.0f / red[0];
  for (int d = threadIdx.x; d < 2 * HH; d += blockDim.x) {
    float acc = 0.0f;
    for (int s = 0; s < SS; ++s)
      acc += logit[s] * hsbuf[((size_t)(b * SS + s)) * (2 * HH) + d];
    pooled[(size_t)b * (2 * HH) + d] = acc * inv;
  }
}

// ---------------- BN over axis 0 for (B, Nfeat) ----------------
__global__ void bn1d(float* __restrict__ X, const float* __restrict__ gamma,
                     const float* __restrict__ beta, int Nfeat) {
  int f = blockIdx.x * blockDim.x + threadIdx.x;
  if (f >= Nfeat) return;
  float a = 0.0f, b2 = 0.0f;
  for (int r = 0; r < BB; ++r) { float v = X[(size_t)r * Nfeat + f]; a += v; b2 += v * v; }
  float m = a / BB;
  float rs = rsqrtf(b2 / BB - m * m + EPSV);
  for (int r = 0; r < BB; ++r) {
    size_t i = (size_t)r * Nfeat + f;
    X[i] = (X[i] - m) * rs * gamma[f] + beta[f];
  }
}

// ---------------- dueling combine ----------------
__global__ void duel_combine(const float* __restrict__ adv, const float* __restrict__ val,
                             float* __restrict__ out) {
  int b = threadIdx.x;
  if (b >= BB) return;
  float m = 0.0f;
  for (int i = 0; i < 6; ++i) m += adv[b * 6 + i];
  m *= (1.0f / 6.0f);
  for (int i = 0; i < 6; ++i) out[b * 6 + i] = val[b] + adv[b * 6 + i] - m;
}

// ---------------- host driver ----------------
struct PB { __bf16* p; int Kp; };

static inline void gemm(const float* A, int lda, PB Bp, const float* bias,
                        float* C, int ldc, int M, int N, int flags, hipStream_t s) {
  int tm = M >> 4, tn4 = (N + 63) >> 6;
  int tiles = tm * tn4;
  int blocks = (tiles + 7) / 8;
  if (blocks > 2048) blocks = 2048;
  if (blocks < 1) blocks = 1;
  gemm_bf16_wmma<<<dim3(blocks), dim3(256), 0, s>>>(A, lda, Bp.p, Bp.Kp, bias, C, ldc, M, N, flags);
}

extern "C" void kernel_launch(void* const* d_in, const int* in_sizes, int n_in,
                              void* d_out, int out_size, void* d_ws, size_t ws_size,
                              hipStream_t stream) {
  // inputs in setup_inputs() dict order
  const float* states_t = (const float*)d_in[0];
  const float* states_a = (const float*)d_in[1];
  const float* states_v = (const float*)d_in[2];
  const float* g_bn_t   = (const float*)d_in[3];
  const float* b_bn_t   = (const float*)d_in[4];
  const float* g_bn_v   = (const float*)d_in[5];
  const float* b_bn_v   = (const float*)d_in[6];
  const float* g_bn_m   = (const float*)d_in[7];
  const float* b_bn_m   = (const float*)d_in[8];
  const float* W_uni    = (const float*)d_in[9];
  const float* b_uni    = (const float*)d_in[10];
  const float* W_uni_v  = (const float*)d_in[11];
  const float* b_uni_v  = (const float*)d_in[12];
  const float* W_bi     = (const float*)d_in[13];
  const float* b_bi     = (const float*)d_in[14];
  const float* W_mul    = (const float*)d_in[15];
  const float* b_mul    = (const float*)d_in[16];
  const float* W_att    = (const float*)d_in[17];
  const float* W_rel    = (const float*)d_in[18];
  const float* W_self   = (const float*)d_in[19];
  const float* b_gnn    = (const float*)d_in[20];
  const float* W1a      = (const float*)d_in[21];
  const float* b1a      = (const float*)d_in[22];
  const float* W2a      = (const float*)d_in[23];
  const float* b2a      = (const float*)d_in[24];
  const float* W1v      = (const float*)d_in[25];
  const float* b1v      = (const float*)d_in[26];
  const float* W2v      = (const float*)d_in[27];
  const float* b2v      = (const float*)d_in[28];
  const float* uWih_f = (const float*)d_in[29];
  const float* uWhh_f = (const float*)d_in[30];
  const float* ubih_f = (const float*)d_in[31];
  const float* ubhh_f = (const float*)d_in[32];
  const float* uWih_b = (const float*)d_in[33];
  const float* uWhh_b = (const float*)d_in[34];
  const float* ubih_b = (const float*)d_in[35];
  const float* ubhh_b = (const float*)d_in[36];
  const float* mWih_f = (const float*)d_in[37];
  const float* mWhh_f = (const float*)d_in[38];
  const float* mbih_f = (const float*)d_in[39];
  const float* mbhh_f = (const float*)d_in[40];
  const float* mWih_b = (const float*)d_in[41];
  const float* mWhh_b = (const float*)d_in[42];
  const float* mbih_b = (const float*)d_in[43];
  const float* mbhh_b = (const float*)d_in[44];

  // workspace partition
  char* wp = (char*)d_ws;
  auto alloc = [&](size_t bytes) -> void* {
    void* p = wp;
    wp += (bytes + 255) & ~(size_t)255;
    return p;
  };
  float* mean   = (float*)alloc(512 * 4);
  float* rstd   = (float*)alloc(512 * 4);
  float* xn     = (float*)alloc((size_t)BB * 512 * SS * 4);     // normalized input (max C=512)
  float* xlin   = (float*)alloc((size_t)BB * SS * HH * 4);
  float* xpf    = (float*)alloc((size_t)BB * SS * 3 * HH * 4);  // fwd GRU projection
  float* xpb    = (float*)alloc((size_t)BB * SS * 3 * HH * 4);  // bwd GRU projection
  float* hsb    = (float*)alloc((size_t)BB * SS * 2 * HH * 4);  // BiGRU output
  float* pool_t = (float*)alloc((size_t)BB * 2 * HH * 4);
  float* pool_a = (float*)alloc((size_t)BB * 2 * HH * 4);
  float* pool_v = (float*)alloc((size_t)BB * 2 * HH * 4);
  float* s_at   = (float*)alloc((size_t)BB * HH * 4);
  float* s_vt   = (float*)alloc((size_t)BB * HH * 4);
  float* avt    = (float*)alloc((size_t)BB * 4 * HH * 4);
  float* msgs   = (float*)alloc((size_t)BB * 4 * 2 * HH * 4);
  float* adv1   = (float*)alloc((size_t)BB * HH * 4);
  float* val1   = (float*)alloc((size_t)BB * HH * 4);
  float* advO   = (float*)alloc((size_t)BB * 16 * 4);
  float* valO   = (float*)alloc((size_t)BB * 16 * 4);

  auto packB = [&](const float* B, int K, int N) -> PB {
    int Kp = (K + 31) & ~31;
    int Np = (N + 63) & ~63;
    __bf16* dst = (__bf16*)alloc((size_t)Kp * Np * 2);
    size_t total = (size_t)Kp * Np;
    int blocks = (int)((total + 255) / 256);
    if (blocks > 2048) blocks = 2048;
    pack_bt<<<dim3(blocks), dim3(256), 0, stream>>>(B, K, N, dst, Kp, Np);
    return PB{dst, Kp};
  };

  // pack all weights once (bf16, transposed, zero-padded)
  PB p_Wuni  = packB(W_uni, FF, HH);
  PB p_Wuniv = packB(W_uni_v, HH, HH);
  PB p_uWihF = packB(uWih_f, HH, 3 * HH);
  PB p_uWihB = packB(uWih_b, HH, 3 * HH);
  PB p_uWhhF = packB(uWhh_f, HH, 3 * HH);
  PB p_uWhhB = packB(uWhh_b, HH, 3 * HH);
  PB p_mWihF = packB(mWih_f, HH, 3 * HH);
  PB p_mWihB = packB(mWih_b, HH, 3 * HH);
  PB p_mWhhF = packB(mWhh_f, HH, 3 * HH);
  PB p_mWhhB = packB(mWhh_b, HH, 3 * HH);
  PB p_Wbi0  = packB(W_bi, 2 * HH, HH);                         // rows 0..1023  (sa/sv part)
  PB p_Wbi1  = packB(W_bi + (size_t)(2 * HH) * HH, 2 * HH, HH); // rows 1024..2047 (st part)
  PB p_Wmul0 = packB(W_mul, HH, 4 * HH);
  PB p_Wmul1 = packB(W_mul + (size_t)HH * 4 * HH, HH, 4 * HH);
  PB p_Wself = packB(W_self, 2 * HH, 2 * HH);
  PB p_W1a   = packB(W1a, 8 * HH, HH);
  PB p_W1v   = packB(W1v, 8 * HH, HH);
  PB p_W2a   = packB(W2a, HH, 6);
  PB p_W2v   = packB(W2v, HH, 1);
  __bf16* p_rel = (__bf16*)alloc((size_t)16 * 2 * HH * 2 * HH * 2);
  for (int blk = 0; blk < 16; ++blk)
    pack_bt<<<dim3(2048), dim3(256), 0, stream>>>(
        W_rel + (size_t)blk * (2 * HH) * (2 * HH), 2 * HH, 2 * HH,
        p_rel + (size_t)blk * (2 * HH) * (2 * HH), 2 * HH, 2 * HH);

  auto runBranch = [&](const float* states, int Cin, PB pW, const float* gbn,
                       const float* bbn, const float* bl, float* pool) {
    bn_stats<<<dim3(Cin), dim3(256), 0, stream>>>(states, Cin, mean, rstd);
    bn_apply<<<dim3(2048), dim3(256), 0, stream>>>(states, mean, rstd, gbn, bbn, xn, Cin);
    // reshape (B,C,S)->(B,S',C') is a pure reinterpretation: xn is contiguous (B*S', C')
    gemm(xn, Cin, pW, bl, xlin, HH, BB * SS, HH, GF_BIAS, stream);
    gemm(xlin, HH, p_uWihF, ubih_f, xpf, 3 * HH, BB * SS, 3 * HH, GF_BIAS, stream);
    gemm(xlin, HH, p_uWihB, ubih_b, xpb, 3 * HH, BB * SS, 3 * HH, GF_BIAS, stream);
    gru_scan_bi<<<dim3(2), dim3(1024), 0, stream>>>(xpf, xpb, p_uWhhF.p, p_uWhhB.p,
                                                    ubhh_f, ubhh_b, hsb, SS);
    attn_pool<<<dim3(BB), dim3(256), 0, stream>>>(hsb, W_att, pool);
  };

  runBranch(states_t, FF, p_Wuni, g_bn_t, b_bn_t, b_uni, pool_t);
  runBranch(states_a, FF, p_Wuni, g_bn_t, b_bn_t, b_uni, pool_a);
  runBranch(states_v, HH, p_Wuniv, g_bn_v, b_bn_v, b_uni_v, pool_v);

  // s_at = [sa|st] @ W_bi + b_bi ; s_vt = [sv|st] @ W_bi + b_bi  (split-K accumulation)
  gemm(pool_a, 2 * HH, p_Wbi0, b_bi, s_at, HH, BB, HH, GF_BIAS, stream);
  gemm(pool_t, 2 * HH, p_Wbi1, nullptr, s_at, HH, BB, HH, GF_ACC, stream);
  gemm(pool_v, 2 * HH, p_Wbi0, b_bi, s_vt, HH, BB, HH, GF_BIAS, stream);
  gemm(pool_t, 2 * HH, p_Wbi1, nullptr, s_vt, HH, BB, HH, GF_ACC, stream);

  // avt = [s_at|s_vt] @ W_mul + b_mul, then BN over batch
  gemm(s_at, HH, p_Wmul0, b_mul, avt, 4 * HH, BB, 4 * HH, GF_BIAS, stream);
  gemm(s_vt, HH, p_Wmul1, nullptr, avt, 4 * HH, BB, 4 * HH, GF_ACC, stream);
  bn1d<<<dim3((4 * HH + 255) / 256), dim3(256), 0, stream>>>(avt, g_bn_m, b_bn_m, 4 * HH);

  // mini BiGRU over 4 steps: avt viewed as (B*4, H)
  gemm(avt, HH, p_mWihF, mbih_f, xpf, 3 * HH, BB * 4, 3 * HH, GF_BIAS, stream);
  gemm(avt, HH, p_mWihB, mbih_b, xpb, 3 * HH, BB * 4, 3 * HH, GF_BIAS, stream);
  gru_scan_bi<<<dim3(2), dim3(1024), 0, stream>>>(xpf, xpb, p_mWhhF.p, p_mWhhB.p,
                                                  mbhh_f, mbhh_b, hsb, 4);
  // hsb now holds out4 = (B, 4, 2H) contiguous = (128, 1024)

  // msgs[b,k,:] = sum_j out4[b,j,:] @ W_rel[j,k]
  for (int k = 0; k < 4; ++k)
    for (int j = 0; j < 4; ++j)
      gemm(hsb + (size_t)j * (2 * HH), 4 * 2 * HH,
           PB{p_rel + (size_t)(j * 4 + k) * (2 * HH) * (2 * HH), 2 * HH},
           nullptr, msgs + (size_t)k * (2 * HH), 4 * 2 * HH,
           BB, 2 * HH, (j == 0) ? 0 : GF_ACC, stream);
  // h = relu(out4 @ W_self + msgs + b_gnn), in place in msgs
  gemm(hsb, 2 * HH, p_Wself, b_gnn, msgs, 2 * HH,
       BB * 4, 2 * HH, GF_ACC | GF_BIAS | GF_RELU, stream);

  // dueling heads: flat = msgs viewed as (B, 8H)
  gemm(msgs, 8 * HH, p_W1a, b1a, adv1, HH, BB, HH, GF_BIAS | GF_RELU, stream);
  gemm(adv1, HH, p_W2a, b2a, advO, 6, BB, 6, GF_BIAS, stream);
  gemm(msgs, 8 * HH, p_W1v, b1v, val1, HH, BB, HH, GF_BIAS | GF_RELU, stream);
  gemm(val1, HH, p_W2v, b2v, valO, 1, BB, 1, GF_BIAS, stream);

  duel_combine<<<dim3(1), dim3(32), 0, stream>>>(advO, valO, (float*)d_out);
  (void)in_sizes; (void)n_in; (void)out_size; (void)ws_size;
}
